// GRU_87256555585692
// MI455X (gfx1250) — matrix operand order repaired and checked by reading, not compiled, not run on hardware
//
#include <hip/hip_runtime.h>

// ---------------------------------------------------------------------------
// GRU encoder/decoder seq2seq for MI455X (gfx1250), wave32 + WMMA bf16.
// Each 256-thread workgroup owns a 32-row batch tile and runs the whole
// recurrence with hidden state resident in LDS. Weights converted once to
// bf16 in d_ws (L2-resident). Gate GEMMs use v_wmma_f32_16x16x32_bf16.
// Each wave owns an N-column strip and computes BOTH M-tiles per B-fragment
// load, so every weight byte crosses L2 exactly once per time step.
// ---------------------------------------------------------------------------

#define H    256
#define G3   768
#define MT   32      // batch rows per workgroup (2 WMMA M-tiles)
#define TENC 15
#define NTHREADS 256 // 8 waves

typedef __attribute__((ext_vector_type(16))) __bf16 v16bf;
typedef __attribute__((ext_vector_type(8)))  __bf16 v8bf;
typedef __attribute__((ext_vector_type(8)))  float  v8f;

union FragA { v16bf v; v8bf h[2]; };

__device__ __forceinline__ float sigmoid_f(float x) {
    return 1.f / (1.f + __expf(-x));
}
__device__ __forceinline__ float tanh_f(float x) {
    return 2.f / (1.f + __expf(-2.f * x)) - 1.f;
}

// ---------------------------------------------------------------------------
// Fused gate GEMM:  rz_sum = in*Wih.T + h*Whh.T  (r,z gates summed),
//                   inn    = in*Wih.T (n gate), hn = h*Whh.T (n gate)
// A: row-major bf16 [MT x H] in LDS.  B: row-major bf16 [G3 x H] in global
// (W.T column n == W row n, so B fragments are contiguous 32B per lane).
// Each wave: one nt strip, both M-tiles -> 4 WMMAs per B-fragment pair.
// ---------------------------------------------------------------------------
__device__ __forceinline__ void gemm_gates(
    const __bf16* __restrict__ sh_in, const __bf16* __restrict__ sh_h,
    const __bf16* __restrict__ Wih,   const __bf16* __restrict__ Whh,
    float* __restrict__ sh_rz, float* __restrict__ sh_inn,
    float* __restrict__ sh_hn, int wid, int lane)
{
    const int lm = lane & 15;
    const int kh = lane >> 4;           // 0 or 1
    for (int nt = wid; nt < G3 / 16; nt += 8) {
        v8f ai0 = {}, ai1 = {}, ah0 = {}, ah1 = {};
        const __bf16* arow_i = sh_in + lm * H + kh * 16;      // M-tile 0
        const __bf16* arow_h = sh_h  + lm * H + kh * 16;
        const __bf16* brow_i = Wih + (nt * 16 + lm) * H + kh * 16;
        const __bf16* brow_h = Whh + (nt * 16 + lm) * H + kh * 16;
#pragma unroll
        for (int k0 = 0; k0 < H; k0 += 32) {
            const v16bf bi = *(const v16bf*)(brow_i + k0);
            const v16bf bh = *(const v16bf*)(brow_h + k0);
            FragA a0, a1;
            // input-path A fragments, M-tiles 0 and 1
            a0.h[0] = *(const v8bf*)(arow_i + k0);
            a0.h[1] = *(const v8bf*)(arow_i + k0 + 8);
            a1.h[0] = *(const v8bf*)(arow_i + 16 * H + k0);
            a1.h[1] = *(const v8bf*)(arow_i + 16 * H + k0 + 8);
            ai0 = __builtin_amdgcn_wmma_f32_16x16x32_bf16(
                false, a0.v, false, bi, (short)0, ai0, false, false);
            ai1 = __builtin_amdgcn_wmma_f32_16x16x32_bf16(
                false, a1.v, false, bi, (short)0, ai1, false, false);
            // hidden-path A fragments
            a0.h[0] = *(const v8bf*)(arow_h + k0);
            a0.h[1] = *(const v8bf*)(arow_h + k0 + 8);
            a1.h[0] = *(const v8bf*)(arow_h + 16 * H + k0);
            a1.h[1] = *(const v8bf*)(arow_h + 16 * H + k0 + 8);
            ah0 = __builtin_amdgcn_wmma_f32_16x16x32_bf16(
                false, a0.v, false, bh, (short)0, ah0, false, false);
            ah1 = __builtin_amdgcn_wmma_f32_16x16x32_bf16(
                false, a1.v, false, bh, (short)0, ah1, false, false);
        }
        // C/D layout: lanes 0-15 -> M=r, lanes 16-31 -> M=8+r; N = lane&15
        const int mb  = kh * 8;
        const int col = nt * 16 + lm;         // gate column 0..767
        if (col < 512) {                      // r,z: sum input+hidden paths
#pragma unroll
            for (int r = 0; r < 8; ++r) {
                sh_rz[(mb + r) * 512 + col]      = ai0[r] + ah0[r];
                sh_rz[(16 + mb + r) * 512 + col] = ai1[r] + ah1[r];
            }
        } else {                              // n: keep paths separate
            const int c = col - 512;
#pragma unroll
            for (int r = 0; r < 8; ++r) {
                sh_inn[(mb + r) * H + c]      = ai0[r];
                sh_hn [(mb + r) * H + c]      = ah0[r];
                sh_inn[(16 + mb + r) * H + c] = ai1[r];
                sh_hn [(16 + mb + r) * H + c] = ah1[r];
            }
        }
    }
}

// Elementwise GRU combine: h' = (1-z)*tanh(inn + r*hn) + z*h
__device__ __forceinline__ void gru_update(
    __bf16* __restrict__ sh_h, const float* __restrict__ sh_rz,
    const float* __restrict__ sh_inn, const float* __restrict__ sh_hn,
    const float* __restrict__ bih, const float* __restrict__ bhh, int tid)
{
    for (int idx = tid; idx < MT * H; idx += NTHREADS) {
        const int m = idx >> 8, j = idx & (H - 1);
        const float r  = sigmoid_f(sh_rz[m * 512 + j] + bih[j] + bhh[j]);
        const float z  = sigmoid_f(sh_rz[m * 512 + 256 + j] + bih[256 + j] + bhh[256 + j]);
        const float nn = tanh_f(sh_inn[idx] + bih[512 + j] +
                                r * (sh_hn[idx] + bhh[512 + j]));
        const float hp = (float)sh_h[idx];
        sh_h[idx] = (__bf16)((1.f - z) * nn + z * hp);
    }
}

// dec_in = relu(h @ We.T + be) -> bf16 LDS; one nt strip per wave, both M-tiles
__device__ __forceinline__ void dec_embed(
    const __bf16* __restrict__ sh_h, __bf16* __restrict__ sh_in,
    const __bf16* __restrict__ We, const float* __restrict__ be,
    int wid, int lane)
{
    const int lm = lane & 15;
    const int kh = lane >> 4;
    for (int nt = wid; nt < H / 16; nt += 8) {
        v8f a0c = {}, a1c = {};
        const __bf16* arow = sh_h + lm * H + kh * 16;
        const __bf16* brow = We   + (nt * 16 + lm) * H + kh * 16;
#pragma unroll
        for (int k0 = 0; k0 < H; k0 += 32) {
            const v16bf b = *(const v16bf*)(brow + k0);
            FragA a0, a1;
            a0.h[0] = *(const v8bf*)(arow + k0);
            a0.h[1] = *(const v8bf*)(arow + k0 + 8);
            a1.h[0] = *(const v8bf*)(arow + 16 * H + k0);
            a1.h[1] = *(const v8bf*)(arow + 16 * H + k0 + 8);
            a0c = __builtin_amdgcn_wmma_f32_16x16x32_bf16(
                false, a0.v, false, b, (short)0, a0c, false, false);
            a1c = __builtin_amdgcn_wmma_f32_16x16x32_bf16(
                false, a1.v, false, b, (short)0, a1c, false, false);
        }
        const int mb = kh * 8;
        const int j  = nt * 16 + lm;
        const float bj = be[j];
#pragma unroll
        for (int r = 0; r < 8; ++r) {
            float v0 = a0c[r] + bj;
            float v1 = a1c[r] + bj;
            sh_in[(mb + r) * H + j]      = (__bf16)(v0 > 0.f ? v0 : 0.f);
            sh_in[(16 + mb + r) * H + j] = (__bf16)(v1 > 0.f ? v1 : 0.f);
        }
    }
}

// ---------------------------------------------------------------------------
// Weight prep: fp32 -> bf16 into workspace (one pass; L2-resident afterwards)
// ---------------------------------------------------------------------------
__global__ void cvt_weights(const float* __restrict__ eWih,
                            const float* __restrict__ eWhh,
                            const float* __restrict__ dWih,
                            const float* __restrict__ dWhh,
                            const float* __restrict__ dWe,
                            __bf16* __restrict__ ws)
{
    const int i = blockIdx.x * blockDim.x + threadIdx.x;
    if (i < G3 * H) {
        ws[0 * G3 * H + i] = (__bf16)eWih[i];
        ws[1 * G3 * H + i] = (__bf16)eWhh[i];
        ws[2 * G3 * H + i] = (__bf16)dWih[i];
        ws[3 * G3 * H + i] = (__bf16)dWhh[i];
    }
    if (i < H * H) ws[4 * G3 * H + i] = (__bf16)dWe[i];
}

// ---------------------------------------------------------------------------
// Main fused seq2seq kernel: one workgroup = 32 batch rows, full recurrence.
// Dynamic LDS layout (160KB total -> 2 WGs per WGP):
//   [0,16K)     sh_h   bf16 [MT][H]
//   [16K,32K)   sh_in  bf16 [MT][H]
//   [32K,96K)   sh_rz  f32  [MT][512]   (x-stage buffer aliases first 512B)
//   [96K,128K)  sh_inn f32  [MT][H]
//   [128K,160K) sh_hn  f32  [MT][H]
// ---------------------------------------------------------------------------
__global__ __launch_bounds__(NTHREADS) void gru_s2s(
    const float* __restrict__ x,
    const float* __restrict__ encWe, const float* __restrict__ encBe,
    const float* __restrict__ encBih, const float* __restrict__ encBhh,
    const float* __restrict__ decBe,
    const float* __restrict__ decBih, const float* __restrict__ decBhh,
    const float* __restrict__ outW, const float* __restrict__ outB,
    const __bf16* __restrict__ wbf, const int* __restrict__ predLenPtr,
    float* __restrict__ out, int N)
{
    extern __shared__ char smem[];
    __bf16* sh_h   = (__bf16*)(smem);
    __bf16* sh_in  = (__bf16*)(smem + 16384);
    float*  sh_rz  = (float*) (smem + 32768);
    float*  sh_inn = (float*) (smem + 98304);
    float*  sh_hn  = (float*) (smem + 131072);
    float*  sh_x   = sh_rz;   // alias: x-stage used only before gemm writes rz

    const int tid  = threadIdx.x;
    const int wid  = tid >> 5;
    const int lane = tid & 31;
    const int b0   = blockIdx.x * MT;
    const int pred = *predLenPtr;

    const __bf16* encWih = wbf;
    const __bf16* encWhh = wbf + 1 * G3 * H;
    const __bf16* decWih = wbf + 2 * G3 * H;
    const __bf16* decWhh = wbf + 3 * G3 * H;
    const __bf16* decWe  = wbf + 4 * G3 * H;

    // h0 = 0
    for (int idx = tid; idx < MT * H; idx += NTHREADS)
        sh_h[idx] = (__bf16)0.f;
    __syncthreads();

    // ---------------- Encoder ----------------
    for (int t = 0; t < TENC; ++t) {
        if (tid < MT * 4) {                       // stage x tile [MT x 4]
            const int m = tid >> 2, c = tid & 3;
            const int n = b0 + m;
            sh_x[tid] = (n < N) ? x[(n * TENC + t) * 4 + c] : 0.f;
        }
        __syncthreads();
        // embed: relu(x @ We.T + be), K=4 -> VALU
        for (int idx = tid; idx < MT * H; idx += NTHREADS) {
            const int m = idx >> 8, j = idx & (H - 1);
            float acc = encBe[j];
#pragma unroll
            for (int c = 0; c < 4; ++c)
                acc += sh_x[m * 4 + c] * encWe[j * 4 + c];
            sh_in[idx] = (__bf16)(acc > 0.f ? acc : 0.f);
        }
        __syncthreads();
        gemm_gates(sh_in, sh_h, encWih, encWhh, sh_rz, sh_inn, sh_hn, wid, lane);
        __syncthreads();
        gru_update(sh_h, sh_rz, sh_inn, sh_hn, encBih, encBhh, tid);
        __syncthreads();
    }

    // dec_in0 = relu(h_enc @ dec_embed_W.T + b)
    dec_embed(sh_h, sh_in, decWe, decBe, wid, lane);
    __syncthreads();

    // ---------------- Decoder (autoregressive) ----------------
    for (int t = 0; t < pred; ++t) {
        gemm_gates(sh_in, sh_h, decWih, decWhh, sh_rz, sh_inn, sh_hn, wid, lane);
        __syncthreads();
        gru_update(sh_h, sh_rz, sh_inn, sh_hn, decBih, decBhh, tid);
        __syncthreads();
        // output head: [MT x 4] = h @ out_W.T + out_b (tiny, VALU)
        if (tid < MT * 4) {
            const int m = tid >> 2, o = tid & 3;
            float acc = outB[o];
            for (int k = 0; k < H; ++k)
                acc += (float)sh_h[m * H + k] * outW[o * H + k];
            const int n = b0 + m;
            if (n < N) out[(n * pred + t) * 4 + o] = acc;
        }
        // next decoder input (concurrent with output head; both read sh_h)
        dec_embed(sh_h, sh_in, decWe, decBe, wid, lane);
        __syncthreads();
    }
}

// ---------------------------------------------------------------------------
extern "C" void kernel_launch(void* const* d_in, const int* in_sizes, int n_in,
                              void* d_out, int out_size, void* d_ws, size_t ws_size,
                              hipStream_t stream)
{
    const float* x       = (const float*)d_in[0];
    const float* encWe   = (const float*)d_in[1];
    const float* encBe   = (const float*)d_in[2];
    const float* encWih  = (const float*)d_in[3];
    const float* encWhh  = (const float*)d_in[4];
    const float* encBih  = (const float*)d_in[5];
    const float* encBhh  = (const float*)d_in[6];
    const float* decWe   = (const float*)d_in[7];
    const float* decBe   = (const float*)d_in[8];
    const float* decWih  = (const float*)d_in[9];
    const float* decWhh  = (const float*)d_in[10];
    const float* decBih  = (const float*)d_in[11];
    const float* decBhh  = (const float*)d_in[12];
    const float* outW    = (const float*)d_in[13];
    const float* outB    = (const float*)d_in[14];
    const int*   predLen = (const int*)d_in[15];

    float* out = (float*)d_out;
    __bf16* wbf = (__bf16*)d_ws;

    const int N = in_sizes[0] / (TENC * 4);

    // 1) fp32 -> bf16 weight conversion into workspace
    cvt_weights<<<(G3 * H + 255) / 256, 256, 0, stream>>>(
        encWih, encWhh, decWih, decWhh, decWe, wbf);

    // 2) fused recurrence, 160KB dynamic LDS per workgroup
    const int blocks = (N + MT - 1) / MT;
    const size_t smem_bytes = 160 * 1024;
    gru_s2s<<<blocks, NTHREADS, smem_bytes, stream>>>(
        x, encWe, encBe, encBih, encBhh, decBe, decBih, decBhh,
        outW, outB, wbf, predLen, out, N);
}
